// callPredictor_58660663329111
// MI455X (gfx1250) — compile-verified, matrix operand order checked by reading, not run on hardware
//
#include <hip/hip_runtime.h>

// Time-aware LSTM, fused single kernel for gfx1250 (MI455X).
// B=4096, TN=256, F=64, H=16, L=1.
// One block = 16 batch rows for the whole sequence; 4 wave32 waves, one per gate.
// X tiles streamed with async global->LDS copies (ASYNCcnt) when available.

#define BQ 4096
#define TNQ 256
#define FF 64
#define HH 16

#define AS1 __attribute__((address_space(1)))
#define AS3 __attribute__((address_space(3)))

#if defined(__has_builtin)
#  if __has_builtin(__builtin_amdgcn_global_load_async_to_lds_b128) && \
      __has_builtin(__builtin_amdgcn_s_wait_asynccnt)
#    define USE_ASYNC_LDS 1
#  endif
#endif
#ifndef USE_ASYNC_LDS
#  define USE_ASYNC_LDS 0
#endif

typedef __attribute__((ext_vector_type(16))) _Float16 v16h;
typedef __attribute__((ext_vector_type(8)))  _Float16 v8h;
typedef __attribute__((ext_vector_type(8)))  float    v8f;
typedef __attribute__((ext_vector_type(4)))  float    v4f;
typedef __attribute__((ext_vector_type(4)))  int      v4i;

typedef AS1 v4i* gvec4p;   // global-AS int4*  (async builtin's source type)
typedef AS3 v4i* lvec4p;   // LDS-AS int4*     (async builtin's dest type)

__device__ __forceinline__ float fast_sigmoid(float x) {
  float e = __builtin_amdgcn_exp2f(-1.4426950408889634f * x);
  return __builtin_amdgcn_rcpf(1.0f + e);
}
__device__ __forceinline__ float fast_tanh(float x) {
  float e = __builtin_amdgcn_exp2f(-2.8853900817779268f * x);
  return 2.0f * __builtin_amdgcn_rcpf(1.0f + e) - 1.0f;
}
__device__ __forceinline__ v16h cat8(v8h lo, v8h hi) {
  return __builtin_shufflevector(lo, hi, 0,1,2,3,4,5,6,7,8,9,10,11,12,13,14,15);
}

__global__ __launch_bounds__(128)
void tlstm_fused(const float* __restrict__ X,  const float* __restrict__ M,
                 const float* __restrict__ cM, const float* __restrict__ T,
                 const float* __restrict__ Wx, const float* __restrict__ bx,
                 const float* __restrict__ Wh, const float* __restrict__ bh,
                 const float* __restrict__ Wc, const float* __restrict__ bc,
                 const float* __restrict__ Wp, const float* __restrict__ bp,
                 float* __restrict__ out_all, float* __restrict__ out_cur)
{
  __shared__ __attribute__((aligned(16))) float    sXf[2][16][FF]; // X tile, f32, double-buffered
  __shared__ __attribute__((aligned(32))) _Float16 sH [16][HH];    // h_t (f16, WMMA A)
  __shared__ __attribute__((aligned(16))) _Float16 sCh[16][HH];    // c_t (f16, WMMA A)
  __shared__ float sC  [16][HH];                                   // c_t (f32)
  __shared__ float sCs [16][HH];                                   // tanh(c@Wc^T+bc)
  __shared__ float sGate[4][16][HH];                               // activated gates
  __shared__ float sTt[2][16];                                     // tanh(T) double-buffered

  const int tid  = threadIdx.x;
  const int g    = tid >> 5;        // wave id == gate id (0:f 1:i 2:o 3:chat)
  const int lane = tid & 31;
  const int n    = lane & 15;       // N column (B/C/D) == M row (A)
  const int hl   = lane >> 4;       // lane half
  const int b0   = blockIdx.x * 16;

  // ---- preload weights as f16 WMMA B-operands (32x16 K-major layout) ----
  v16h wxB0, wxB1, whB, wcB;
  {
    const float* wr = Wx + (16*g + n) * FF;
    #pragma unroll
    for (int j = 0; j < 8; ++j) {
      wxB0[j]   = (_Float16)wr[      8*hl + j];
      wxB0[8+j] = (_Float16)wr[16 +  8*hl + j];
      wxB1[j]   = (_Float16)wr[32 +  8*hl + j];
      wxB1[8+j] = (_Float16)wr[48 +  8*hl + j];
    }
    const float* whr = Wh + (16*g + n) * HH;   // K=16, zero-padded to 32
    const float* wcr = Wc + n * HH;
    #pragma unroll
    for (int j = 0; j < 8; ++j) {
      whB[j] = (_Float16)whr[8*hl + j];  whB[8+j] = (_Float16)0.0f;
      wcB[j] = (_Float16)wcr[8*hl + j];  wcB[8+j] = (_Float16)0.0f;
    }
  }
  const float gateBias = bx[16*g + n] + bh[16*g + n];
  const float cBias    = bc[n];
  const float bp0      = bp[0];

  // projection weights for wave-2 low lanes
  float wp[HH]; float curacc = 0.0f;
  if (g == 2 && lane < 16) {
    #pragma unroll
    for (int j = 0; j < HH; ++j) wp[j] = Wp[j];
  }

  // zero recurrent state
  for (int e = tid; e < 256; e += 128) {
    int m = e >> 4, c = e & 15;
    sH[m][c] = (_Float16)0.0f; sCh[m][c] = (_Float16)0.0f; sC[m][c] = 0.0f;
  }

  // X staging: each thread owns 8 consecutive floats (two 16B chunks) of the tile
  const int mld = tid >> 3;
  const int fld = (tid & 7) * 8;
  const float* xrow0 = X + ((size_t)(b0 + mld) * TNQ) * FF + fld;

#if USE_ASYNC_LDS
#define ISSUE_X(tt) do {                                                        \
    const float* p_ = xrow0 + (size_t)(tt) * FF;                                \
    float* l_ = &sXf[(tt) & 1][mld][fld];                                       \
    __builtin_amdgcn_global_load_async_to_lds_b128((gvec4p)p_,       (lvec4p)l_,       0, 0); \
    __builtin_amdgcn_global_load_async_to_lds_b128((gvec4p)(p_ + 4), (lvec4p)(l_ + 4), 0, 0); \
  } while (0)
  ISSUE_X(0);                        // prologue: start t=0 copy
#endif

  for (int t = 0; t < TNQ; ++t) {
    // ---- Phase A: make X tile for step t resident in LDS ----
#if USE_ASYNC_LDS
    __builtin_amdgcn_s_wait_asynccnt(0);      // tile t landed in sXf[t&1]
#else
    {
      const float* p = xrow0 + (size_t)t * FF;
      v4f x0 = *(const v4f*)p;
      v4f x1 = *(const v4f*)(p + 4);
      *(v4f*)&sXf[t & 1][mld][fld]     = x0;
      *(v4f*)&sXf[t & 1][mld][fld + 4] = x1;
      if (t + 1 < TNQ) __builtin_prefetch(p + FF, 0, 0);
    }
#endif
    if (tid < 16) sTt[t & 1][tid] = fast_tanh(T[(b0 + tid) * TNQ + t]);
    __syncthreads();   // B1

#if USE_ASYNC_LDS
    if (t + 1 < TNQ) ISSUE_X(t + 1);  // overlap t+1 copy with step-t compute
#endif

    // ---- Phase B: gate = X@Wx^T(slice) + h@Wh^T(slice) + bias via WMMA ----
    v8f acc;
    #pragma unroll
    for (int r = 0; r < 8; ++r) acc[r] = gateBias;

    const float* xr = &sXf[t & 1][n][0];      // A row = lane%16
    v16h a0, a1;
    #pragma unroll
    for (int j = 0; j < 8; ++j) {             // f32 LDS -> f16 A operands
      a0[j]   = (_Float16)xr[      8*hl + j];
      a0[8+j] = (_Float16)xr[16 +  8*hl + j];
      a1[j]   = (_Float16)xr[32 +  8*hl + j];
      a1[8+j] = (_Float16)xr[48 +  8*hl + j];
    }
    v8h z8 = {};
    v16h a2 = cat8(*(const v8h*)&sH[n][8*hl], z8);   // h, K padded to 32

    acc = __builtin_amdgcn_wmma_f32_16x16x32_f16(false, a0, false, wxB0, (short)0, acc, false, false);
    acc = __builtin_amdgcn_wmma_f32_16x16x32_f16(false, a1, false, wxB1, (short)0, acc, false, false);
    acc = __builtin_amdgcn_wmma_f32_16x16x32_f16(false, a2, false, whB,  (short)0, acc, false, false);

    if (g == 0) {  // c_s = tanh(c@Wc^T + bc)
      v8f accC;
      #pragma unroll
      for (int r = 0; r < 8; ++r) accC[r] = cBias;
      v16h aC = cat8(*(const v8h*)&sCh[n][8*hl], z8);
      accC = __builtin_amdgcn_wmma_f32_16x16x32_f16(false, aC, false, wcB, (short)0, accC, false, false);
      #pragma unroll
      for (int r = 0; r < 8; ++r) sCs[r + 8*hl][n] = fast_tanh(accC[r]);
    }

    // activations -> sGate (D layout: row = r + 8*hl, col = n)
    #pragma unroll
    for (int r = 0; r < 8; ++r) {
      float v = acc[r];
      v = (g == 3) ? fast_tanh(v) : fast_sigmoid(v);
      sGate[g][r + 8*hl][n] = v;
    }

    // output projection of h_{t-1}, overlapped with other waves' WMMA
    if (g == 2 && lane < 16 && t > 0) {
      v16h hrow = *(const v16h*)&sH[lane][0];
      float dot = bp0;
      #pragma unroll
      for (int j = 0; j < HH; ++j) dot += (float)hrow[j] * wp[j];
      int idx = (b0 + lane) * TNQ + (t - 1);
      float ov = dot * M[idx];
      out_all[idx] = ov;
      curacc += ov * cM[idx];
    }
    __syncthreads();   // B2

    // ---- Phase C: elementwise recurrence (2 elements / thread) ----
    #pragma unroll
    for (int rep = 0; rep < 2; ++rep) {
      int e = tid + rep * 128;
      int m = e >> 4, c = e & 15;
      float fg = sGate[0][m][c], ig = sGate[1][m][c];
      float og = sGate[2][m][c], cg = sGate[3][m][c];
      float ct = sC[m][c], cs = sCs[m][c];
      float tt = sTt[t & 1][m];
      float cstar = ct + cs * (tt - 1.0f);      // (c - c_s) + c_s*tanh(T)
      float cnew  = fg * cstar + ig * cg;       // cg already tanh'd
      float hnew  = og * fast_tanh(cnew);
      sC [m][c] = cnew;
      sCh[m][c] = (_Float16)cnew;
      sH [m][c] = (_Float16)hnew;
    }
  }

  __syncthreads();
  // epilogue: project h_{TN-1}, write masked time-sum
  if (g == 2 && lane < 16) {
    v16h hrow = *(const v16h*)&sH[lane][0];
    float dot = bp0;
    #pragma unroll
    for (int j = 0; j < HH; ++j) dot += (float)hrow[j] * wp[j];
    int idx = (b0 + lane) * TNQ + (TNQ - 1);
    float ov = dot * M[idx];
    out_all[idx] = ov;
    curacc += ov * cM[idx];
    out_cur[b0 + lane] = curacc;
  }
}

extern "C" void kernel_launch(void* const* d_in, const int* in_sizes, int n_in,
                              void* d_out, int out_size, void* d_ws, size_t ws_size,
                              hipStream_t stream) {
  const float* X  = (const float*)d_in[0];
  const float* M  = (const float*)d_in[1];
  const float* cM = (const float*)d_in[2];
  const float* T  = (const float*)d_in[3];
  const float* Wx = (const float*)d_in[4];
  const float* bx = (const float*)d_in[5];
  const float* Wh = (const float*)d_in[6];
  const float* bh = (const float*)d_in[7];
  const float* Wc = (const float*)d_in[8];
  const float* bc = (const float*)d_in[9];
  const float* Wp = (const float*)d_in[10];
  const float* bp = (const float*)d_in[11];
  float* out_all = (float*)d_out;                        // (B,TN,1) flat
  float* out_cur = (float*)d_out + (size_t)BQ * TNQ;     // (B,1)
  (void)in_sizes; (void)n_in; (void)out_size; (void)d_ws; (void)ws_size;

  dim3 grid(BQ / 16), block(128);
  hipLaunchKernelGGL(tlstm_fused, grid, block, 0, stream,
                     X, M, cM, T, Wx, bx, Wh, bh, Wc, bc, Wp, bp, out_all, out_cur);
}
